// Network_7335804141688
// MI455X (gfx1250) — compile-verified, matrix operand order checked by reading, not run on hardware
//
#include <hip/hip_runtime.h>
#include <hip/hip_bf16.h>
#include <math.h>

typedef __attribute__((ext_vector_type(16))) _Float16 v16h;
typedef __attribute__((ext_vector_type(8)))  _Float16 v8h;
typedef __attribute__((ext_vector_type(8)))  float    v8f;

#define T_IN   8192
#define BATCH  32
#define T_OUT  8188
#define HID    96
#define G4     384          // 4*HID
#define KCOMB  128          // 32 (input) + 96 (hidden)

// fast sigmoid/tanh built on v_exp_f32 + v_rcp_f32 (no IEEE divide sequence)
static __device__ __forceinline__ float fast_sigmoid(float x) {
  return __builtin_amdgcn_rcpf(1.f + __expf(-x));
}
static __device__ __forceinline__ float fast_tanh(float x) {
  return 2.f * __builtin_amdgcn_rcpf(1.f + __expf(-2.f * x)) - 1.f;
}

// ---- fragment loader: row-major [rows][ld] f16, tile (row0, k0), K-depth 32 ----
// A-matrix 16x32 f16 layout (ISA 7.12.2): lane half (lane>>4) selects K sub-block,
// lane&15 selects M row. B for D=A*B loads identically from row-major [N][K].
static __device__ __forceinline__ v16h load_frag(const _Float16* base, int ld,
                                                 int row0, int k0, int lane) {
  const int half = lane >> 4;
  const int r    = lane & 15;
  const _Float16* p = base + (size_t)(row0 + r) * ld + k0 + half * 8;
  v8h lo = *(const v8h*)(p);
  v8h hi = *(const v8h*)(p + 16);
  v16h a;
#pragma unroll
  for (int i = 0; i < 8; ++i) { a[i] = lo[i]; a[i + 8] = hi[i]; }
  return a;
}

// =============== Phase 1: fused conv1(1->16,k3) + conv2(16->32,k3) -> x f16 ===============
__global__ void __launch_bounds__(256)
conv_kernel(const float* __restrict__ in, const float* __restrict__ w1,
            const float* __restrict__ b1, const float* __restrict__ w2,
            const float* __restrict__ b2, _Float16* __restrict__ xout) {
  __shared__ float sw1[48], sb1[16], sw2[1536], sb2[32];
  const int tid = threadIdx.x;
  for (int i = tid; i < 48;   i += 256) sw1[i] = w1[i];
  for (int i = tid; i < 16;   i += 256) sb1[i] = b1[i];
  for (int i = tid; i < 1536; i += 256) sw2[i] = w2[i];
  for (int i = tid; i < 32;   i += 256) sb2[i] = b2[i];
  __syncthreads();

  const int idx = blockIdx.x * 256 + tid;
  if (idx >= T_OUT * BATCH) return;
  const int t = idx / BATCH;
  const int b = idx % BATCH;
  const float* ip = in + (size_t)b * T_IN + t;
  const float x0 = ip[0], x1 = ip[1], x2 = ip[2], x3 = ip[3], x4 = ip[4];

  float c1[16][3];
#pragma unroll
  for (int ci = 0; ci < 16; ++ci) {
    const float k0 = sw1[ci*3+0], k1 = sw1[ci*3+1], k2 = sw1[ci*3+2], bb = sb1[ci];
    c1[ci][0] = bb + k0*x0 + k1*x1 + k2*x2;
    c1[ci][1] = bb + k0*x1 + k1*x2 + k2*x3;
    c1[ci][2] = bb + k0*x2 + k1*x3 + k2*x4;
  }
  _Float16* xo = xout + (size_t)t * (BATCH * 32) + b * 32;
#pragma unroll
  for (int co = 0; co < 32; ++co) {
    float acc = sb2[co];
    const float* wp = sw2 + co * 48;
#pragma unroll
    for (int ci = 0; ci < 16; ++ci)
      acc += wp[ci*3+0]*c1[ci][0] + wp[ci*3+1]*c1[ci][1] + wp[ci*3+2]*c1[ci][2];
    xo[co] = (_Float16)acc;
  }
}

// =============== Phase 2: pack [w_ih | w_hh] -> f16 [384][128], bias sum, lin_w f16 ===============
__global__ void prep_kernel(const float* __restrict__ w_ih, const float* __restrict__ w_hh,
                            const float* __restrict__ b_ih, const float* __restrict__ b_hh,
                            const float* __restrict__ lin_w,
                            _Float16* __restrict__ Wc, float* __restrict__ biasc,
                            _Float16* __restrict__ lw) {
  const int tid = threadIdx.x;
  if (tid < G4) {
    for (int c = 0; c < 32;  ++c) Wc[tid * KCOMB + c]      = (_Float16)w_ih[tid * 32 + c];
    for (int c = 0; c < HID; ++c) Wc[tid * KCOMB + 32 + c] = (_Float16)w_hh[tid * HID + c];
    biasc[tid] = b_ih[tid] + b_hh[tid];
  }
  if (tid < 128) {
    for (int c = 0; c < HID; ++c) lw[tid * HID + c] = (_Float16)lin_w[tid * HID + c];
  }
}

// =============== Phase 3: sequential LSTM, 2 blocks (one 16-row batch tile each) ===============
// 6 waves/block; wave w owns hidden cols [16w,16w+16): computes its i/f/g/o tiles,
// keeps c in accumulator-layout registers, writes h (f16) to LDS + global history.
// Bias lives in persistent v8f fragments and feeds the FIRST WMMA of each gate
// chain as its C operand (D != C), so no per-step re-splat copies are needed.
__global__ void __launch_bounds__(192, 1)
lstm_kernel(const _Float16* __restrict__ x, const _Float16* __restrict__ Wc,
            const float* __restrict__ biasc, _Float16* __restrict__ hall) {
  __shared__ __align__(16) _Float16 Abuf[16 * KCOMB];   // [m][ x(32) | h(96) ]

  const int tid  = threadIdx.x;
  const int wave = tid >> 5;
  const int lane = tid & 31;
  const int half = lane >> 4;
  const int r    = lane & 15;
  const int hn   = wave * 16;            // hidden column tile
  const int m0g  = blockIdx.x * 16;      // global batch-row base for this block

  // zero entire A buffer (h part must start at 0)
  for (int i = tid; i < 16 * KCOMB; i += 192) Abuf[i] = (_Float16)0.f;

  // resident recurrent-weight fragments: Bf[gate][kchunk]
  v16h Bf[4][4];
#pragma unroll
  for (int g = 0; g < 4; ++g)
#pragma unroll
    for (int kc = 0; kc < 4; ++kc)
      Bf[g][kc] = load_frag(Wc, KCOMB, g * HID + hn, kc * 32, lane);

  // persistent bias fragments (C/D layout: bias depends only on N = hn + r)
  v8f bfrag[4];
#pragma unroll
  for (int g = 0; g < 4; ++g) {
    const float b = biasc[g * HID + hn + r];
#pragma unroll
    for (int i = 0; i < 8; ++i) bfrag[g][i] = b;
  }

  v8f cst = {0.f, 0.f, 0.f, 0.f, 0.f, 0.f, 0.f, 0.f};

  for (int t = 0; t < T_OUT; ++t) {
    // stage x_t tile [16][32] into Abuf cols 0..31 (256 dwords)
    const unsigned int* xs = (const unsigned int*)(x + (size_t)t * (BATCH * 32) + m0g * 32);
    for (int j = tid; j < 256; j += 192) {
      unsigned int v = xs[j];
      *(unsigned int*)&Abuf[(j >> 4) * KCOMB + (j & 15) * 2] = v;
    }
    if (t + 1 < T_OUT)
      __builtin_prefetch(x + (size_t)(t + 1) * (BATCH * 32) + m0g * 32 + (tid & 15) * 64, 0, 0);
    __syncthreads();

    v8f acc[4];
    // kc = 0 peeled: C input is the persistent bias fragment (no accumulator init)
    {
      v16h a0 = load_frag(Abuf, KCOMB, 0, 0, lane);
#pragma unroll
      for (int g = 0; g < 4; ++g)
        acc[g] = __builtin_amdgcn_wmma_f32_16x16x32_f16(
            false, a0, false, Bf[g][0], (short)0, bfrag[g], false, false);
    }
#pragma unroll
    for (int kc = 1; kc < 4; ++kc) {
      v16h a = load_frag(Abuf, KCOMB, 0, kc * 32, lane);
#pragma unroll
      for (int g = 0; g < 4; ++g)
        acc[g] = __builtin_amdgcn_wmma_f32_16x16x32_f16(
            false, a, false, Bf[g][kc], (short)0, acc[g], false, false);
    }

    // gate nonlinearities + state update (i,f,g,o order) — fast rcp, no IEEE divide
    v8f hv;
#pragma unroll
    for (int i = 0; i < 8; ++i) {
      const float ig = fast_sigmoid(acc[0][i]);
      const float fg = fast_sigmoid(acc[1][i]);
      const float gg = fast_tanh(acc[2][i]);
      const float og = fast_sigmoid(acc[3][i]);
      const float c  = fg * cst[i] + ig * gg;
      cst[i] = c;
      hv[i]  = og * fast_tanh(c);
    }

    __syncthreads();   // all waves done reading Abuf before h is overwritten

    _Float16* hg = hall + (size_t)t * (BATCH * HID);
#pragma unroll
    for (int i = 0; i < 8; ++i) {
      const int m = i + half * 8;                 // local batch row (C/D layout)
      const _Float16 h16 = (_Float16)hv[i];
      Abuf[m * KCOMB + 32 + hn + r] = h16;        // feed next step
      hg[(size_t)(m0g + m) * HID + hn + r] = h16; // history for output GEMM
    }
  }
}

// =============== Phase 4: out[T*B,128] = h_all[T*B,96] * lin_w^T + lin_b ===============
__global__ void __launch_bounds__(256)
out_kernel(const _Float16* __restrict__ hall, const _Float16* __restrict__ lw,
           const float* __restrict__ lin_b, float* __restrict__ out) {
  const int tid  = threadIdx.x;
  const int lane = tid & 31;
  const int half = lane >> 4;
  const int r    = lane & 15;
  const int wid  = blockIdx.x * 8 + (tid >> 5);
  const int m0   = (wid >> 3) * 16;      // row tile (T*B = 262016 = 16376*16 exactly)
  const int n0   = (wid & 7) * 16;       // 128 output cols = 8 tiles

  const float b = lin_b[n0 + r];
  v8f acc = {b, b, b, b, b, b, b, b};
#pragma unroll
  for (int kc = 0; kc < 3; ++kc) {
    v16h a  = load_frag(hall, HID, m0, kc * 32, lane);
    v16h bf = load_frag(lw,   HID, n0, kc * 32, lane);
    acc = __builtin_amdgcn_wmma_f32_16x16x32_f16(false, a, false, bf, (short)0, acc,
                                                 false, false);
  }
#pragma unroll
  for (int i = 0; i < 8; ++i)
    out[(size_t)(m0 + i + half * 8) * 128 + n0 + r] = acc[i];
}

// ======================================================================================
extern "C" void kernel_launch(void* const* d_in, const int* in_sizes, int n_in,
                              void* d_out, int out_size, void* d_ws, size_t ws_size,
                              hipStream_t stream) {
  const float* input  = (const float*)d_in[0];
  const float* c1w    = (const float*)d_in[1];
  const float* c1b    = (const float*)d_in[2];
  const float* c2w    = (const float*)d_in[3];
  const float* c2b    = (const float*)d_in[4];
  const float* w_ih   = (const float*)d_in[5];
  const float* w_hh   = (const float*)d_in[6];
  const float* b_ih   = (const float*)d_in[7];
  const float* b_hh   = (const float*)d_in[8];
  const float* lin_w  = (const float*)d_in[9];
  const float* lin_b  = (const float*)d_in[10];

  char* ws = (char*)d_ws;
  const size_t OFF_X  = 0;                                        // f16 [8188][32][32]  ~16 MB
  const size_t OFF_H  = OFF_X + (size_t)T_OUT * BATCH * 32 * 2;   // f16 [8188][32][96] ~48 MB
  const size_t OFF_W  = OFF_H + (size_t)T_OUT * BATCH * HID * 2;  // f16 [384][128]
  const size_t OFF_B  = OFF_W + (size_t)G4 * KCOMB * 2;           // f32 [384]
  const size_t OFF_LW = OFF_B + (size_t)G4 * 4;                   // f16 [128][96]

  _Float16* x16   = (_Float16*)(ws + OFF_X);
  _Float16* hall  = (_Float16*)(ws + OFF_H);
  _Float16* Wc    = (_Float16*)(ws + OFF_W);
  float*    biasc = (float*)   (ws + OFF_B);
  _Float16* lw    = (_Float16*)(ws + OFF_LW);

  conv_kernel<<<(T_OUT * BATCH + 255) / 256, 256, 0, stream>>>(input, c1w, c1b, c2w, c2b, x16);
  prep_kernel<<<1, 384, 0, stream>>>(w_ih, w_hh, b_ih, b_hh, lin_w, Wc, biasc, lw);
  lstm_kernel<<<2, 192, 0, stream>>>(x16, Wc, biasc, hall);
  out_kernel<<<16376, 256, 0, stream>>>(hall, lw, lin_b, (float*)d_out);
}
